// CrossEntropyGuidedPolicy_73890617360625
// MI455X (gfx1250) — matrix operand order repaired and checked by reading, not compiled, not run on hardware
//
#include <hip/hip_runtime.h>
#include <hip/hip_bf16.h>

typedef __attribute__((ext_vector_type(16))) _Float16 v16h;
typedef __attribute__((ext_vector_type(8)))  float    v8f;

#define ACTION_DIM 32
#define ITERS      4
#define CEM_BATCH  64
#define TOPK       6
#define STATE_DIM  64
#define HIDDEN     128
#define ENVS_PER_BLOCK 16
#define THREADS    (ENVS_PER_BLOCK * 32)

// ---------- counter-based RNG: PCG hash + paired Box-Muller ----------
__device__ __forceinline__ unsigned pcg(unsigned x) {
    x = x * 747796405u + 2891336453u;
    unsigned w = ((x >> ((x >> 28u) + 4u)) ^ x) * 277803737u;
    return (w >> 22u) ^ w;
}
// two N(0,1) values for dim pair (dimPair, dimPair+1); dimPair must be even
__device__ __forceinline__ float2 gauss2(unsigned it, unsigned env,
                                         unsigned cem, unsigned dimPair) {
    unsigned s  = pcg((env * 131071u) ^ (cem * 2654435761u) ^
                      (dimPair * 40503u) ^ (it * 0x9E3779B9u));
    unsigned h1 = pcg(s);
    unsigned h2 = pcg(h1 ^ 0x85ebca6bu);
    float u1 = 1.0f - (float)(h1 >> 8) * (1.0f / 16777216.0f);  // (0,1]
    float u2 =        (float)(h2 >> 8) * (1.0f / 16777216.0f);  // [0,1)
    float r   = sqrtf(fmaxf(-2.0f * __logf(u1), 0.0f));
    float ang = 6.28318530718f * u2;
    return make_float2(r * __cosf(ang), r * __sinf(ang));
}

__device__ __forceinline__ v8f wmma16x16x32(v16h a, v16h b, v8f c) {
    return __builtin_amdgcn_wmma_f32_16x16x32_f16(
        false, a, false, b, (short)0, c, false, false);
}

__global__ __launch_bounds__(THREADS, 1)
void cem_policy_kernel(const float* __restrict__ state,
                       const float* __restrict__ W1,
                       const float* __restrict__ b1,
                       const float* __restrict__ W2,
                       float* __restrict__ out,
                       int nenv) {
    __shared__ float     state_lds[ENVS_PER_BLOCK][STATE_DIM];   // 4 KB
    __shared__ float     S_lds[ENVS_PER_BLOCK][HIDDEN];          // 8 KB
    __shared__ _Float16  Bh[8][32][16];                          // 8 KB, W1_a B-operands
    __shared__ float     mean_lds[ENVS_PER_BLOCK][ACTION_DIM];   // 2 KB
    __shared__ float     std_lds[ENVS_PER_BLOCK][ACTION_DIM];    // 2 KB
    __shared__ float     Q_lds[ENVS_PER_BLOCK][CEM_BATCH];       // 4 KB

    const int tid  = threadIdx.x;
    const int wave = tid >> 5;
    const int lane = tid & 31;
    const int l15  = lane & 15;
    const int hi   = lane >> 4;         // 0: lanes 0-15, 1: lanes 16-31
    const int kb   = hi * 8;            // A-operand K base per lane half
    const int env_base = blockIdx.x * ENVS_PER_BLOCK;

    __builtin_prefetch(W1 + tid * 24, 0, 3);   // warm caches for weights

    // ---- async-stage the block's state tile (16 envs x 64 f32 = 4KB) -----
    // threads 0..255 (waves 0..7) each move one 16-byte chunk, Memory -> LDS
    if (wave < 8) {
        long fmax = (long)nenv * STATE_DIM - 4;
        long fidx = (long)env_base * STATE_DIM + (long)tid * 4;
        if (fidx > fmax) fidx = fmax;
        const float* gaddr = state + fidx;
        unsigned lds_off = (unsigned)(uintptr_t)&state_lds[0][0] + (unsigned)tid * 16u;
        asm volatile("global_load_async_to_lds_b128 %0, %1, off"
                     :: "v"(lds_off), "v"(gaddr) : "memory");
    } else {
        // ---- waves 8..15: build W1_a f16 B-operands in LDS ----------------
        const int nt = wave - 8;
        const float* p = W1 + (size_t)64 * HIDDEN + nt * 16 + l15;
        v16h bv;
#pragma unroll
        for (int j = 0; j < 16; ++j) bv[j] = (_Float16)p[(size_t)(kb + (j & 7) + ((j >= 8) ? 16 : 0)) * HIDDEN];
        *(v16h*)&Bh[nt][lane][0] = bv;
    }
    asm volatile("s_wait_asynccnt 0x0" ::: "memory");
    __syncthreads();

    // ======= Phase 1: S = state @ W1[0:64] + b1  (waves 0..7) =============
    if (wave < 8) {
        const int ntile = wave;
        v16h b0, bv1;                  // B: lane = N column, 16 K in table order
        {
            const float* p0 = W1 + ntile * 16 + l15;
            const float* p1 = W1 + (size_t)32 * HIDDEN + ntile * 16 + l15;
#pragma unroll
            for (int j = 0; j < 16; ++j) {
                size_t kr = (size_t)(kb + (j & 7) + ((j >= 8) ? 16 : 0)) * HIDDEN;
                b0[j]  = (_Float16)p0[kr];
                bv1[j] = (_Float16)p1[kr];
            }
        }
        v16h a0, a1;                   // A: 16 envs x 64 state dims (2 chunks)
        {
            const float* sp = &state_lds[l15][0];
#pragma unroll
            for (int v = 0; v < 8; ++v) {
                int kk = kb + ((v >= 4) ? 16 : 0) + 2 * (v & 3);
                a0[2*v]   = (_Float16)sp[kk];
                a0[2*v+1] = (_Float16)sp[kk + 1];
                a1[2*v]   = (_Float16)sp[32 + kk];
                a1[2*v+1] = (_Float16)sp[32 + kk + 1];
            }
        }
        v8f acc = {};
        acc = wmma16x16x32(a0, b0,  acc);
        acc = wmma16x16x32(a1, bv1, acc);
        float bb = b1[ntile * 16 + l15];
#pragma unroll
        for (int r = 0; r < 8; ++r) {
            int m = hi * 8 + r;                       // env-local row (C layout)
            S_lds[m][ntile * 16 + l15] = acc[r] + bb;
        }
    }
    __syncthreads();
    // ====== everything below is wave-local: wave `wave` owns one env ======

    const int envL = wave;
    const unsigned env = (unsigned)(env_base + wave);
    if ((int)env >= nenv) return;

    float mean_r = 0.0f, std_r = 1.0f;      // lane d owns dim d
    mean_lds[envL][lane] = 0.0f;
    std_lds[envL][lane]  = 1.0f;

    float s_row[8], w2v[8];                 // per-lane epilogue constants
#pragma unroll
    for (int nt = 0; nt < 8; ++nt) {
        s_row[nt] = S_lds[envL][nt * 16 + l15];
        w2v[nt]   = W2[nt * 16 + l15];
    }

    for (unsigned it = 0; it < ITERS; ++it) {
        // -- gather mean/std for this lane's K set (contiguous pairs) -------
        float mk[16], sk[16];
#pragma unroll
        for (int v = 0; v < 8; ++v) {
            int kk = kb + ((v >= 4) ? 16 : 0) + 2 * (v & 3);
            mk[2*v]   = mean_lds[envL][kk];
            mk[2*v+1] = mean_lds[envL][kk + 1];
            sk[2*v]   = std_lds[envL][kk];
            sk[2*v+1] = std_lds[envL][kk + 1];
        }
        // -- generate actions directly in WMMA A-operand layout -------------
        v16h A[4];
#pragma unroll
        for (int mt = 0; mt < 4; ++mt) {
            unsigned cem = (unsigned)(mt * 16 + l15);
#pragma unroll
            for (int v = 0; v < 8; ++v) {
                int kk = kb + ((v >= 4) ? 16 : 0) + 2 * (v & 3);
                float2 g = gauss2(it, env, cem, (unsigned)kk);
                A[mt][2*v]   = (_Float16)fmaf(sk[2*v],   g.x, mk[2*v]);
                A[mt][2*v+1] = (_Float16)fmaf(sk[2*v+1], g.y, mk[2*v+1]);
            }
        }
        // -- Q = sum_j relu(S[j] + (A@W1_a)[cem,j]) * W2[j] ------------------
        float qp[4][8];
#pragma unroll
        for (int mt = 0; mt < 4; ++mt)
#pragma unroll
            for (int r = 0; r < 8; ++r) qp[mt][r] = 0.0f;

#pragma unroll
        for (int nt = 0; nt < 8; ++nt) {
            v16h b = *(const v16h*)&Bh[nt][lane][0];
            float sv = s_row[nt], wv = w2v[nt];
#pragma unroll
            for (int mt = 0; mt < 4; ++mt) {
                v8f c = {};
                c = wmma16x16x32(A[mt], b, c);
#pragma unroll
                for (int r = 0; r < 8; ++r)
                    qp[mt][r] = fmaf(fmaxf(c[r] + sv, 0.0f), wv, qp[mt][r]);
            }
        }
#pragma unroll
        for (int mt = 0; mt < 4; ++mt) {
#pragma unroll
            for (int r = 0; r < 8; ++r) {        // reduce over N (16 lanes)
                float q = qp[mt][r];
                q += __shfl_xor(q, 1, 32);
                q += __shfl_xor(q, 2, 32);
                q += __shfl_xor(q, 4, 32);
                q += __shfl_xor(q, 8, 32);
                if (l15 == 0) Q_lds[envL][mt * 16 + hi * 8 + r] = q;
            }
        }
        // -- top-6 over 64 Q values (2 per lane) ----------------------------
        float q_a = Q_lds[envL][lane];
        float q_b = Q_lds[envL][lane + 32];
        bool alive_a = true, alive_b = true;
        int Is[TOPK];
#pragma unroll
        for (int k = 0; k < TOPK; ++k) {
            float va = alive_a ? q_a : -3.4e38f;
            float vb = alive_b ? q_b : -3.4e38f;
            float bv; int bi;
            if (va >= vb) { bv = va; bi = lane; } else { bv = vb; bi = lane + 32; }
#pragma unroll
            for (int off = 16; off >= 1; off >>= 1) {
                float ov = __shfl_xor(bv, off, 32);
                int   oi = __shfl_xor(bi, off, 32);
                if (ov > bv || (ov == bv && oi < bi)) { bv = ov; bi = oi; }
            }
            Is[k] = bi;
            if (bi == lane)      alive_a = false;
            if (bi == lane + 32) alive_b = false;
        }
        // -- regenerate selected actions, update mean/std (ddof=1) ----------
        float sum = 0.0f, sum2 = 0.0f, best = 0.0f;
        unsigned dpair = (unsigned)(lane & ~1);
#pragma unroll
        for (int k = 0; k < TOPK; ++k) {
            float2 g = gauss2(it, env, (unsigned)Is[k], dpair);
            float e = (lane & 1) ? g.y : g.x;
            float a = fmaf(std_r, e, mean_r);
            if (k == 0) best = a;
            sum  += a;
            sum2 = fmaf(a, a, sum2);
        }
        float nm  = sum * (1.0f / TOPK);
        float var = (sum2 - (float)TOPK * nm * nm) * (1.0f / (TOPK - 1));
        float ns  = sqrtf(fmaxf(var, 0.0f));
        mean_r = nm; std_r = ns;
        mean_lds[envL][lane] = nm;
        std_lds[envL][lane]  = ns;

        if (it == ITERS - 1)
            out[(size_t)env * ACTION_DIM + lane] = best;
    }
}

extern "C" void kernel_launch(void* const* d_in, const int* in_sizes, int n_in,
                              void* d_out, int out_size, void* d_ws, size_t ws_size,
                              hipStream_t stream) {
    const float* state = (const float*)d_in[0];  // (B_ENV, 64)
    const float* W1    = (const float*)d_in[1];  // (96, 128)
    const float* b1    = (const float*)d_in[2];  // (128,)
    const float* W2    = (const float*)d_in[3];  // (128, 1)
    // d_in[4] = b2: additive constant on Q, irrelevant to top-k ordering.
    float* out = (float*)d_out;                  // (B_ENV, 32)

    int nenv = in_sizes[0] / STATE_DIM;
    int blocks = (nenv + ENVS_PER_BLOCK - 1) / ENVS_PER_BLOCK;
    cem_policy_kernel<<<blocks, THREADS, 0, stream>>>(state, W1, b1, W2, out, nenv);
}
